// singleSelfAttention_60954175864968
// MI455X (gfx1250) — matrix operand order repaired and checked by reading, not compiled, not run on hardware
//
#include <hip/hip_runtime.h>
#include <hip/hip_bf16.h>

typedef __attribute__((ext_vector_type(16))) __bf16 v16bf;
typedef __attribute__((ext_vector_type(8)))  float  v8f;
typedef __attribute__((ext_vector_type(8)))  unsigned v8u;

__device__ __forceinline__ unsigned short f2bf(float f) {
    unsigned u = __builtin_bit_cast(unsigned, f);
    unsigned r = (u + 0x7FFFu + ((u >> 16) & 1u)) >> 16;   // round-to-nearest-even
    return (unsigned short)r;
}

__device__ __forceinline__ v16bf pack16(uint4 a, uint4 b) {
    v8u u;
    u[0] = a.x; u[1] = a.y; u[2] = a.z; u[3] = a.w;
    u[4] = b.x; u[5] = b.y; u[6] = b.z; u[7] = b.w;
    return __builtin_bit_cast(v16bf, u);
}

// Async DMA of 16 bytes global -> LDS (per lane). vdst holds the LDS byte address,
// vaddr the 64-bit global address (GV mode). Tracked by ASYNCcnt.
__device__ __forceinline__ void async_ld_b128(unsigned ldsOff, const void* gaddr) {
    asm volatile("global_load_async_to_lds_b128 %0, %1, off"
                 :: "v"(ldsOff), "v"((unsigned long long)(size_t)gaddr)
                 : "memory");
}
__device__ __forceinline__ void wait_asynccnt0() {
    asm volatile("s_wait_asynccnt 0x0" ::: "memory");
}

// ---------------- cast kernels ----------------
__global__ __launch_bounds__(256) void cast_f32_bf16_flat(const float* __restrict__ in,
                                                          unsigned short* __restrict__ out) {
    int i = blockIdx.x * 256 + threadIdx.x;            // each thread: 4 elements
    float4 v = ((const float4*)in)[i];
    uint2 o;
    o.x = ((unsigned)f2bf(v.y) << 16) | f2bf(v.x);
    o.y = ((unsigned)f2bf(v.w) << 16) | f2bf(v.z);
    ((uint2*)out)[i] = o;
}

__global__ __launch_bounds__(256) void transpose_cast_bf16(const float* __restrict__ in,
                                                           unsigned short* __restrict__ out) {
    int idx = blockIdx.x * 256 + threadIdx.x;          // in is [1024 x 1024] row-major
    int i = idx >> 10, o = idx & 1023;
    out[(size_t)o * 1024 + i] = f2bf(in[idx]);
}

// ---------------- WMMA GEMM: C = A * B, A bf16, B supplied as B^T row-major (N x K) -----
// Block tile 128(M) x 256(N), BK = 32 (one bf16 WMMA K-step). 256 threads = 8 waves
// (wave32), wave grid 2(M) x 4(N); each wave owns 64x64 = 4x4 fragments -> 16 WMMAs per
// k-step. LDS double-buffered; the NEXT k-tile is DMA'd global->LDS with
// global_load_async_to_lds_b128 (no VGPR staging, no spills) while WMMAs run, then
// s_wait_asynccnt 0 + barrier.
#define LDSP 40   // padded LDS row stride in bf16 elements (80 bytes, 16B aligned)

template <bool OBF16, bool OTRANS>
__global__ __launch_bounds__(256) void wmma_gemm_bt(
        const unsigned short* __restrict__ A, const unsigned short* __restrict__ Bt,
        void* __restrict__ Cptr, int Kd, int lda, int ldb, int ldc,
        long long aBatch, long long bBatch, long long cBatch, float scale) {
    __shared__ unsigned short sA[2][128 * LDSP];
    __shared__ unsigned short sB[2][256 * LDSP];

    const int z = blockIdx.z;
    const unsigned short* Ab = A  + (size_t)z * aBatch;
    const unsigned short* Bb = Bt + (size_t)z * bBatch;

    const int t     = threadIdx.x;
    const int lane  = t & 31;
    const int wave  = t >> 5;
    const int waveM = wave >> 2;       // 0..1
    const int waveN = wave & 3;        // 0..3
    const int l15   = lane & 15;
    const int grp   = lane >> 4;

    // staging assignment: 2 threads per row, 16 contiguous bf16 each (two b128 DMAs)
    const int ar = t >> 1;
    const int ac = (t & 1) * 16;
    const size_t aRowG = (size_t)blockIdx.y * 128 + ar;   // M row
    const size_t bRowG = (size_t)blockIdx.x * 256 + ar;   // N rows of B^T (two passes)

    const unsigned aLds0 = (unsigned)(size_t)&sA[0][ar * LDSP + ac];
    const unsigned aLds1 = (unsigned)(size_t)&sA[1][ar * LDSP + ac];
    const unsigned bLds0 = (unsigned)(size_t)&sB[0][ar * LDSP + ac];
    const unsigned bLds1 = (unsigned)(size_t)&sB[1][ar * LDSP + ac];
    const unsigned bStep = 128 * LDSP * 2;                // byte offset rows 128..255

    auto stageAsync = [&](int kt, int buf) {
        const int k0 = kt << 5;
        const unsigned aL = buf ? aLds1 : aLds0;
        const unsigned bL = buf ? bLds1 : bLds0;
        const unsigned short* ga = Ab + aRowG * lda + k0 + ac;
        async_ld_b128(aL,      ga);
        async_ld_b128(aL + 16, ga + 8);
        const unsigned short* gb0 = Bb + bRowG * ldb + k0 + ac;
        async_ld_b128(bL,      gb0);
        async_ld_b128(bL + 16, gb0 + 8);
        const unsigned short* gb1 = gb0 + (size_t)128 * ldb;
        async_ld_b128(bL + bStep,      gb1);
        async_ld_b128(bL + bStep + 16, gb1 + 8);
    };

    v8f acc[4][4] = {};

    const int ktiles = Kd >> 5;
    stageAsync(0, 0);
    wait_asynccnt0();
    __syncthreads();

    for (int kt = 0; kt < ktiles; ++kt) {
        const int cur = kt & 1;
        if (kt + 1 < ktiles) stageAsync(kt + 1, cur ^ 1);  // DMA next tile during WMMAs
        if (kt + 2 < ktiles) {                             // prefetch k+2 into caches
            const int k2 = (kt + 2) << 5;
            __builtin_prefetch((const char*)(Ab + aRowG * lda + k2 + ac), 0, 1);
            __builtin_prefetch((const char*)(Bb + bRowG * ldb + k2 + ac), 0, 1);
        }

        // ---- fragment loads, contiguous k = 16*grp + s for both A and B ----
        v16bf afr[4], bfr[4];
#pragma unroll
        for (int m = 0; m < 4; ++m) {
            int row = waveM * 64 + m * 16 + l15;
            const uint4* p = (const uint4*)&sA[cur][row * LDSP + grp * 16];
            afr[m] = pack16(p[0], p[1]);     // contiguous K=[grp*16,+16)
        }
#pragma unroll
        for (int n = 0; n < 4; ++n) {
            int col = waveN * 64 + n * 16 + l15;
            const uint4* p = (const uint4*)&sB[cur][col * LDSP + grp * 16];
            bfr[n] = pack16(p[0], p[1]);     // contiguous K=[grp*16,+16)
        }
#pragma unroll
        for (int m = 0; m < 4; ++m)
#pragma unroll
            for (int n = 0; n < 4; ++n)
                acc[m][n] = __builtin_amdgcn_wmma_f32_16x16x32_bf16(
                    false, afr[m], false, bfr[n], (short)0, acc[m][n], false, false);

        wait_asynccnt0();                  // our DMAs into the other buffer are done
        __syncthreads();                   // everyone's are
    }

    // ---- store C (f32 or bf16, optionally transposed) ----
    const size_t cz = (size_t)z * cBatch;
#pragma unroll
    for (int m = 0; m < 4; ++m) {
#pragma unroll
        for (int n = 0; n < 4; ++n) {
            int fr = blockIdx.y * 128 + waveM * 64 + m * 16;
            int fc = blockIdx.x * 256 + waveN * 64 + n * 16 + l15;
#pragma unroll
            for (int r = 0; r < 8; ++r) {
                int row = fr + r + grp * 8;
                float v = acc[m][n][r] * scale;
                size_t idx = OTRANS ? ((size_t)fc * ldc + row) : ((size_t)row * ldc + fc);
                if (OBF16) ((unsigned short*)Cptr)[cz + idx] = f2bf(v);
                else       ((float*)Cptr)[cz + idx] = v;
            }
        }
    }
}

// ------- row softmax: f32 logits in, bf16 probabilities out (256 thr / 2048-wide row) ---
__global__ __launch_bounds__(256) void softmax_rows(const float* __restrict__ S,
                                                    unsigned short* __restrict__ P) {
    __shared__ float red[256];
    const float* p = S + (size_t)blockIdx.x * 2048;
    unsigned short* q = P + (size_t)blockIdx.x * 2048;
    int t = threadIdx.x;
    float v[8], mx = -3.4e38f;
#pragma unroll
    for (int i = 0; i < 8; ++i) { v[i] = p[t + i * 256]; mx = fmaxf(mx, v[i]); }
    red[t] = mx; __syncthreads();
    for (int s = 128; s > 0; s >>= 1) { if (t < s) red[t] = fmaxf(red[t], red[t + s]); __syncthreads(); }
    mx = red[0]; __syncthreads();
    float sum = 0.f;
#pragma unroll
    for (int i = 0; i < 8; ++i) { v[i] = __expf(v[i] - mx); sum += v[i]; }
    red[t] = sum; __syncthreads();
    for (int s = 128; s > 0; s >>= 1) { if (t < s) red[t] += red[t + s]; __syncthreads(); }
    float inv = 1.0f / red[0];
#pragma unroll
    for (int i = 0; i < 8; ++i) q[t + i * 256] = f2bf(v[i] * inv);
}

// ---------------- host-side orchestration ----------------
extern "C" void kernel_launch(void* const* d_in, const int* in_sizes, int n_in,
                              void* d_out, int out_size, void* d_ws, size_t ws_size,
                              hipStream_t stream) {
    const float* y  = (const float*)d_in[0];   // [4,2048,1024]
    const float* Wq = (const float*)d_in[1];   // [1024,1024] (in x out)
    const float* Wk = (const float*)d_in[2];
    const float* Wv = (const float*)d_in[3];
    float* out = (float*)d_out;                // [4,2048,1024]

    char* ws = (char*)d_ws;
    unsigned short* yb  = (unsigned short*)(ws + 0);            // 16 MB  y bf16
    unsigned short* wqt = (unsigned short*)(ws + 16777216);     //  2 MB  Wq^T bf16 (out x in)
    unsigned short* wkt = (unsigned short*)(ws + 18874368);
    unsigned short* wvt = (unsigned short*)(ws + 20971520);
    unsigned short* qb  = (unsigned short*)(ws + 23068672);     // 16 MB  Q bf16
    unsigned short* kb  = (unsigned short*)(ws + 39845888);     // 16 MB  K bf16
    unsigned short* vt  = (unsigned short*)(ws + 56623104);     // 16 MB  V^T bf16 [4][1024][2048]
    float*          Sb  = (float*)(ws + 73400320);              // 64 MB  logits f32 [4][2048][2048]
    unsigned short* Pb  = (unsigned short*)(ws + 140509184);    // 32 MB  probs bf16 [4][2048][2048]

    // 1. casts
    cast_f32_bf16_flat<<<8192, 256, 0, stream>>>(y, yb);                // 8M elems / 4
    transpose_cast_bf16<<<4096, 256, 0, stream>>>(Wq, wqt);
    transpose_cast_bf16<<<4096, 256, 0, stream>>>(Wk, wkt);
    transpose_cast_bf16<<<4096, 256, 0, stream>>>(Wv, wvt);

    const long long SEQ = 2048, DIN = 1024;
    // 2. projections (M=2048/batch, N=1024, K=1024)
    dim3 gp(4, 16, 4);
    wmma_gemm_bt<true, false><<<gp, 256, 0, stream>>>(
        yb, wqt, qb, 1024, 1024, 1024, 1024, SEQ * DIN, 0LL, SEQ * DIN, 1.0f);
    wmma_gemm_bt<true, false><<<gp, 256, 0, stream>>>(
        yb, wkt, kb, 1024, 1024, 1024, 1024, SEQ * DIN, 0LL, SEQ * DIN, 1.0f);
    wmma_gemm_bt<true, true><<<gp, 256, 0, stream>>>(           // V stored transposed
        yb, wvt, vt, 1024, 1024, 1024, 2048, SEQ * DIN, 0LL, DIN * SEQ, 1.0f);

    // 3. S = Q K^T / sqrt(1024)  (M=N=2048, K=1024), f32 out
    dim3 gs(8, 16, 4);
    wmma_gemm_bt<false, false><<<gs, 256, 0, stream>>>(
        qb, kb, Sb, 1024, 1024, 1024, 2048, SEQ * DIN, SEQ * DIN, SEQ * SEQ, 0.03125f);

    // 4. softmax rows -> bf16 probabilities
    softmax_rows<<<8192, 256, 0, stream>>>(Sb, Pb);

    // 5. O = P V  (M=2048, N=1024, K=2048)
    dim3 go(4, 16, 4);
    wmma_gemm_bt<false, false><<<go, 256, 0, stream>>>(
        Pb, vt, out, 2048, 2048, 2048, 1024, SEQ * SEQ, DIN * SEQ, SEQ * DIN, 1.0f);

    (void)in_sizes; (void)n_in; (void)out_size; (void)ws_size;
}
